// CausalSelfAttention_2250562863923
// MI455X (gfx1250) — matrix operand order repaired and checked by reading, not compiled, not run on hardware
//
#include <hip/hip_runtime.h>

#define B_  2
#define T_  2048
#define C_  1024
#define H_  16
#define HD_ 64
#define N3_ 3072

typedef __attribute__((ext_vector_type(16))) _Float16 v16h;
typedef __attribute__((ext_vector_type(8)))  float    v8f;
typedef unsigned int u32x4 __attribute__((ext_vector_type(4)));
typedef int          i32x4 __attribute__((ext_vector_type(4)));
typedef int          i32x8 __attribute__((ext_vector_type(8)));

#if defined(__has_builtin)
#if __has_builtin(__builtin_amdgcn_tensor_load_to_lds)
#define HAVE_TDM 1
#endif
#endif
#if __has_include(<hip/amd_detail/amd_gfx1250_TDM.h>)
#define TDM_ARITY6 1
#endif

union FragH { v16h v; float4 f4[2]; };

__device__ __forceinline__ v8f wmma_f16(v16h a, v16h b, v8f c) {
  // D = A(16x32 f16) * B(32x16 f16) + C(16x16 f32)
  return __builtin_amdgcn_wmma_f32_16x16x32_f16(
      false, a, false, b, (short)0, c, false, false);
}

// A fragment 16x32 f16 from row-major source (row stride ld elements).
// lanes 0-15: row M=lane, K={0..7,16..23}; lanes 16-31: row M=lane-16, K={8..15,24..31}.
__device__ __forceinline__ v16h load_a16x32(const _Float16* base, int ld) {
  const int lane = threadIdx.x & 31;
  const int row  = lane & 15;
  const int kh   = (lane >> 4) << 3;          // 0 or 8
  const _Float16* p = base + (size_t)row * ld + kh;
  FragH u;
  u.f4[0] = *reinterpret_cast<const float4*>(p);       // K = kh .. kh+7
  u.f4[1] = *reinterpret_cast<const float4*>(p + 16);  // K = kh+16 .. kh+23
  return u.v;
}

// B fragment 32x16 f16; source stores each COLUMN contiguously: src[n*ld + k].
// lanes 0-15: col N=lane, K={0..7,16..23}; lanes 16-31: col N=lane-16, K={8..15,24..31}.
__device__ __forceinline__ v16h load_b32x16(const _Float16* base, int ld) {
  const int lane = threadIdx.x & 31;
  const int col  = lane & 15;
  const int kh   = (lane >> 4) << 3;          // 0 or 8
  const _Float16* p = base + (size_t)col * ld + kh;
  FragH u;
  u.f4[0] = *reinterpret_cast<const float4*>(p);
  u.f4[1] = *reinterpret_cast<const float4*>(p + 16);
  return u.v;
}

__device__ __forceinline__ unsigned lds_offset(const void* p) {
  // generic LDS pointer: addr[31:0] is the LDS byte offset (ISA aperture rule)
  return (unsigned)(unsigned long long)p;
}

#ifdef HAVE_TDM
// One-shot TDM 2D tile load: tile_h rows of tile_w f16 elements, row stride
// `stride` elements, into LDS at byte offset lds_off. Issued per-wave (EXEC
// ignored); caller waits on TENSORcnt.
__device__ __forceinline__ void tdm_load_tile_f16(
    unsigned lds_off, const _Float16* gsrc,
    unsigned tile_w, unsigned tile_h, unsigned stride) {
  unsigned long long ga = (unsigned long long)(const void*)gsrc;
  u32x4 g0;
  g0[0] = 1u;                                            // count=1 (valid D#)
  g0[1] = lds_off;                                       // lds_addr
  g0[2] = (unsigned)ga;                                  // global_addr[31:0]
  g0[3] = (unsigned)((ga >> 32) & 0x01FFFFFFu)           // global_addr[56:32]
        | (2u << 30);                                    // type=2 ("image")
  i32x8 g1;
  g1[0] = (int)(1u << 16);                               // data_size=1 -> 2 bytes
  g1[1] = (int)((stride & 0xFFFFu) << 16);               // tensor_dim0[15:0]
  g1[2] = (int)(((stride >> 16) & 0xFFFFu)               // tensor_dim0[31:16]
        | ((tile_h & 0xFFFFu) << 16));                   // tensor_dim1[15:0]
  g1[3] = (int)((tile_w & 0xFFFFu) << 16);               // tile_dim0 (dim1 hi=0)
  g1[4] = (int)(tile_h & 0xFFFFu);                       // tile_dim1 (tile_dim2=0)
  g1[5] = (int)stride;                                   // tensor_dim0_stride lo32
  g1[6] = 0;                                             // stride0 hi / stride1 lo
  g1[7] = 0;
  i32x4 z4 = {0, 0, 0, 0};
#ifdef TDM_ARITY6
  i32x8 z8 = {0, 0, 0, 0, 0, 0, 0, 0};
  __builtin_amdgcn_tensor_load_to_lds(g0, g1, z4, z4, z8, 0);
#else
  __builtin_amdgcn_tensor_load_to_lds(g0, g1, z4, z4, 0);
#endif
}
#endif

// Stage a 16 x C_ f16 A-panel into LDS (TDM if available, else cooperative copy).
__device__ __forceinline__ void stage_a_panel(_Float16* As, const _Float16* gsrc) {
#ifdef HAVE_TDM
  if ((threadIdx.x >> 5) == 0) {
    tdm_load_tile_f16(lds_offset(As), gsrc, C_, 16, C_);
    __builtin_amdgcn_s_wait_tensorcnt(0);
  }
#else
  const float4* src = (const float4*)gsrc;
  float4* dst = (float4*)As;
  for (int i = threadIdx.x; i < (16 * C_) / 8; i += 256) dst[i] = src[i];
#endif
  __syncthreads();
}

// ---------------- precision-convert / transpose prep ----------------
__global__ void k_f32_to_f16(const float* __restrict__ s, _Float16* __restrict__ d, int n) {
  int i = blockIdx.x * blockDim.x + threadIdx.x;
  int stride = gridDim.x * blockDim.x;
  for (; i < n; i += stride) d[i] = (_Float16)s[i];
}

// d[n*K + k] = (f16) s[k*N + n]
__global__ void k_transpose_f16(const float* __restrict__ s, _Float16* __restrict__ d, int K, int N) {
  int i = blockIdx.x * blockDim.x + threadIdx.x;
  int stride = gridDim.x * blockDim.x;
  int total = K * N;
  for (; i < total; i += stride) {
    int n = i / K;
    int k = i - n * K;
    d[i] = (_Float16)s[(size_t)k * N + n];
  }
}

// ---------------- QKV projection: LDS-staged A panel, pipelined K loop ----------------
__global__ __launch_bounds__(256) void k_qkv_gemm(
    const _Float16* __restrict__ xh, const _Float16* __restrict__ wT,
    const float* __restrict__ bias,
    _Float16* __restrict__ Qs, _Float16* __restrict__ Kh, _Float16* __restrict__ Vt) {
  __shared__ __align__(16) _Float16 As[16 * C_];        // 32 KB, shared by 8 waves
  const int lane = threadIdx.x & 31;
  const int wave = threadIdx.x >> 5;
  const int m0 = blockIdx.x * 16;
  const int n0 = (blockIdx.y * 8 + wave) * 64;

  stage_a_panel(As, xh + (size_t)m0 * C_);

  const _Float16* Bb = wT + (size_t)n0 * C_;
  v8f acc[4] = {};
  v16h a = load_a16x32(As, C_);
  v16h b0 = load_b32x16(Bb + 0 * 16 * C_, C_);
  v16h b1 = load_b32x16(Bb + 1 * 16 * C_, C_);
  v16h b2 = load_b32x16(Bb + 2 * 16 * C_, C_);
  v16h b3 = load_b32x16(Bb + 3 * 16 * C_, C_);
  for (int k0 = 0; k0 < C_; k0 += 32) {
    const int kn = (k0 + 32) & (C_ - 1);                // wraps harmlessly on last iter
    v16h an = load_a16x32(As + kn, C_);
    v16h c0 = load_b32x16(Bb + 0 * 16 * C_ + kn, C_);
    v16h c1 = load_b32x16(Bb + 1 * 16 * C_ + kn, C_);
    v16h c2 = load_b32x16(Bb + 2 * 16 * C_ + kn, C_);
    v16h c3 = load_b32x16(Bb + 3 * 16 * C_ + kn, C_);
    acc[0] = wmma_f16(a, b0, acc[0]);
    acc[1] = wmma_f16(a, b1, acc[1]);
    acc[2] = wmma_f16(a, b2, acc[2]);
    acc[3] = wmma_f16(a, b3, acc[3]);
    a = an; b0 = c0; b1 = c1; b2 = c2; b3 = c3;
  }

  const int hi = lane >> 4, nlo = lane & 15;
  const int sel = n0 / C_;                 // 0=Q 1=K 2=V (uniform per strip)
  const int h = (n0 % C_) / HD_;
#pragma unroll
  for (int t = 0; t < 4; ++t) {
    const int d = 16 * t + nlo;
    const float bc = bias[n0 + 16 * t + nlo];
#pragma unroll
    for (int j = 0; j < 8; ++j) {
      const int row = m0 + j + 8 * hi;
      const int b = row >> 11;
      const int tt = row & (T_ - 1);
      const float v = acc[t][j] + bc;
      const size_t hb = (size_t)(b * H_ + h);
      if (sel == 0)      Qs[(hb * T_ + tt) * HD_ + d] = (_Float16)(v * 0.125f);
      else if (sel == 1) Kh[(hb * T_ + tt) * HD_ + d] = (_Float16)v;
      else               Vt[(hb * HD_ + d) * T_ + tt] = (_Float16)v;
    }
  }
}

// ---------------- flash attention: one 16-row Q tile per wave, pipelined ----------------
__global__ __launch_bounds__(256) void k_flash_attn(
    const _Float16* __restrict__ Qs, const _Float16* __restrict__ Kh,
    const _Float16* __restrict__ Vt, _Float16* __restrict__ yh) {
  __shared__ __align__(16) _Float16 Pl[8][16 * 32];     // per-wave P staging
  const int lane = threadIdx.x & 31;
  const int wave = threadIdx.x >> 5;
  const int h = blockIdx.y, b = blockIdx.z;
  const int q0 = (blockIdx.x * 8 + wave) * 16;
  const size_t bh = (size_t)(b * H_ + h);
  const _Float16* Qb = Qs + (bh * T_ + q0) * HD_;
  const _Float16* Kb = Kh + bh * T_ * HD_;
  const _Float16* Vb = Vt + bh * HD_ * T_;
  _Float16* Pw = &Pl[wave][0];

  const v16h aq0 = load_a16x32(Qb, HD_);
  const v16h aq1 = load_a16x32(Qb + 32, HD_);
  v8f o[4] = {};
  float m8[8], l8[8];
#pragma unroll
  for (int j = 0; j < 8; ++j) { m8[j] = -1e30f; l8[j] = 0.f; }
  const int hi = lane >> 4, nlo = lane & 15;
  const int send = q0 + 16;

  // preload K fragments for chunk 0
  v16h kA0 = load_b32x16(Kb, HD_);
  v16h kA1 = load_b32x16(Kb + 32, HD_);
  v16h kB0 = load_b32x16(Kb + (size_t)16 * HD_, HD_);
  v16h kB1 = load_b32x16(Kb + (size_t)16 * HD_ + 32, HD_);

  for (int s0 = 0; s0 < send; s0 += 32) {
    // scores for the two 16-wide key subtiles
    v8f z0 = {}; z0 = wmma_f16(aq0, kA0, z0);
    v8f z1 = {}; z1 = wmma_f16(aq0, kB0, z1);
    v8f sc[2];
    sc[0] = wmma_f16(aq1, kA1, z0);
    sc[1] = wmma_f16(aq1, kB1, z1);

    // prefetch next chunk's K and this chunk's V; latency hidden by softmax VALU
    const int sn = (s0 + 32 < send) ? s0 + 32 : s0;
    kA0 = load_b32x16(Kb + (size_t)sn * HD_, HD_);
    kA1 = load_b32x16(Kb + (size_t)sn * HD_ + 32, HD_);
    kB0 = load_b32x16(Kb + (size_t)(sn + 16) * HD_, HD_);
    kB1 = load_b32x16(Kb + (size_t)(sn + 16) * HD_ + 32, HD_);
    v16h bv0 = load_b32x16(Vb + (size_t)(0 * 16) * T_ + s0, T_);
    v16h bv1 = load_b32x16(Vb + (size_t)(1 * 16) * T_ + s0, T_);
    v16h bv2 = load_b32x16(Vb + (size_t)(2 * 16) * T_ + s0, T_);
    v16h bv3 = load_b32x16(Vb + (size_t)(3 * 16) * T_ + s0, T_);

    // causal mask
#pragma unroll
    for (int sub = 0; sub < 2; ++sub) {
      const int sg = s0 + 16 * sub + nlo;
#pragma unroll
      for (int j = 0; j < 8; ++j) {
        const int rg = q0 + j + 8 * hi;
        if (sg > rg) sc[sub][j] = -1e30f;
      }
    }
    // online softmax (row = 16 lanes of a half-wave)
#pragma unroll
    for (int j = 0; j < 8; ++j) {
      float mx = fmaxf(sc[0][j], sc[1][j]);
#pragma unroll
      for (int off = 8; off > 0; off >>= 1) mx = fmaxf(mx, __shfl_xor(mx, off, 32));
      const float mnew = fmaxf(m8[j], mx);
      const float alpha = __expf(m8[j] - mnew);
      const float p0 = __expf(sc[0][j] - mnew);
      const float p1 = __expf(sc[1][j] - mnew);
      float ls = p0 + p1;
#pragma unroll
      for (int off = 8; off > 0; off >>= 1) ls += __shfl_xor(ls, off, 32);
      l8[j] = l8[j] * alpha + ls;
      m8[j] = mnew;
      sc[0][j] = p0; sc[1][j] = p1;
#pragma unroll
      for (int t = 0; t < 4; ++t) o[t][j] *= alpha;
    }
    // C-layout -> A-layout transpose of P through LDS
#pragma unroll
    for (int sub = 0; sub < 2; ++sub)
#pragma unroll
      for (int j = 0; j < 8; ++j)
        Pw[(j + 8 * hi) * 32 + 16 * sub + nlo] = (_Float16)sc[sub][j];
    asm volatile("s_wait_dscnt 0x0" ::: "memory");       // same-wave LDS RAW fence
    v16h pf = load_a16x32(Pw, 32);
    o[0] = wmma_f16(pf, bv0, o[0]);
    o[1] = wmma_f16(pf, bv1, o[1]);
    o[2] = wmma_f16(pf, bv2, o[2]);
    o[3] = wmma_f16(pf, bv3, o[3]);
  }
#pragma unroll
  for (int t = 0; t < 4; ++t)
#pragma unroll
    for (int j = 0; j < 8; ++j) {
      const int rg = q0 + j + 8 * hi;
      yh[((size_t)b * T_ + rg) * C_ + h * HD_ + 16 * t + nlo] = (_Float16)(o[t][j] / l8[j]);
    }
}

// ---------------- output projection ----------------
__global__ __launch_bounds__(256) void k_out_gemm(
    const _Float16* __restrict__ yh, const _Float16* __restrict__ woT,
    const float* __restrict__ bo, float* __restrict__ out) {
  __shared__ __align__(16) _Float16 As[16 * C_];
  const int lane = threadIdx.x & 31;
  const int wave = threadIdx.x >> 5;
  const int m0 = blockIdx.x * 16;
  const int n0 = (blockIdx.y * 8 + wave) * 64;

  stage_a_panel(As, yh + (size_t)m0 * C_);

  const _Float16* Bb = woT + (size_t)n0 * C_;
  v8f acc[4] = {};
  v16h a = load_a16x32(As, C_);
  v16h b0 = load_b32x16(Bb + 0 * 16 * C_, C_);
  v16h b1 = load_b32x16(Bb + 1 * 16 * C_, C_);
  v16h b2 = load_b32x16(Bb + 2 * 16 * C_, C_);
  v16h b3 = load_b32x16(Bb + 3 * 16 * C_, C_);
  for (int k0 = 0; k0 < C_; k0 += 32) {
    const int kn = (k0 + 32) & (C_ - 1);
    v16h an = load_a16x32(As + kn, C_);
    v16h c0 = load_b32x16(Bb + 0 * 16 * C_ + kn, C_);
    v16h c1 = load_b32x16(Bb + 1 * 16 * C_ + kn, C_);
    v16h c2 = load_b32x16(Bb + 2 * 16 * C_ + kn, C_);
    v16h c3 = load_b32x16(Bb + 3 * 16 * C_ + kn, C_);
    acc[0] = wmma_f16(a, b0, acc[0]);
    acc[1] = wmma_f16(a, b1, acc[1]);
    acc[2] = wmma_f16(a, b2, acc[2]);
    acc[3] = wmma_f16(a, b3, acc[3]);
    a = an; b0 = c0; b1 = c1; b2 = c2; b3 = c3;
  }
  const int hi = lane >> 4, nlo = lane & 15;
#pragma unroll
  for (int t = 0; t < 4; ++t) {
    const int col = n0 + 16 * t + nlo;
    const float bb = bo[col];
#pragma unroll
    for (int j = 0; j < 8; ++j) {
      const int row = m0 + j + 8 * hi;
      out[(size_t)row * C_ + col] = acc[t][j] + bb;
    }
  }
}

extern "C" void kernel_launch(void* const* d_in, const int* in_sizes, int n_in,
                              void* d_out, int out_size, void* d_ws, size_t ws_size,
                              hipStream_t stream) {
  const float* x    = (const float*)d_in[0];
  const float* Wqkv = (const float*)d_in[1];
  const float* bqkv = (const float*)d_in[2];
  const float* Wo   = (const float*)d_in[3];
  const float* bo   = (const float*)d_in[4];
  float* out = (float*)d_out;

  char* ws = (char*)d_ws;
  _Float16* xh    = (_Float16*)ws; ws += (size_t)B_ * T_ * C_ * 2;
  _Float16* wqkvT = (_Float16*)ws; ws += (size_t)N3_ * C_ * 2;
  _Float16* woT   = (_Float16*)ws; ws += (size_t)C_ * C_ * 2;
  _Float16* Qs    = (_Float16*)ws; ws += (size_t)B_ * H_ * T_ * HD_ * 2;
  _Float16* Kh    = (_Float16*)ws; ws += (size_t)B_ * H_ * T_ * HD_ * 2;
  _Float16* Vt    = (_Float16*)ws; ws += (size_t)B_ * H_ * T_ * HD_ * 2;
  _Float16* yh    = (_Float16*)ws;

  k_f32_to_f16<<<dim3(2048), dim3(256), 0, stream>>>(x, xh, B_ * T_ * C_);
  k_transpose_f16<<<dim3(2048), dim3(256), 0, stream>>>(Wqkv, wqkvT, C_, N3_);
  k_transpose_f16<<<dim3(1024), dim3(256), 0, stream>>>(Wo, woT, C_, C_);

  k_qkv_gemm<<<dim3((B_ * T_) / 16, N3_ / 512), dim3(256), 0, stream>>>(
      xh, wqkvT, bqkv, Qs, Kh, Vt);

  k_flash_attn<<<dim3(T_ / 128, H_, B_), dim3(256), 0, stream>>>(Qs, Kh, Vt, yh);

  k_out_gemm<<<dim3((B_ * T_) / 16, C_ / 512), dim3(256), 0, stream>>>(yh, woT, bo, out);
}
